// FaceBlock_80573586472987
// MI455X (gfx1250) — matrix operand order repaired and checked by reading, not compiled, not run on hardware
//
#include <hip/hip_runtime.h>
#include <hip/hip_bf16.h>

typedef _Float16 h16 __attribute__((ext_vector_type(16)));
typedef _Float16 h8  __attribute__((ext_vector_type(8)));
typedef float    f8  __attribute__((ext_vector_type(8)));
typedef int      vi4 __attribute__((__vector_size__(4 * sizeof(int))));

#define B_  2
#define T_  16
#define S_  256
#define NM_ 32
#define C_  3072
#define H_  24
#define D_  128

// ---------------- async global -> LDS (CDNA5), with guarded fallback ----------------

#if defined(__has_builtin)
#if __has_builtin(__builtin_amdgcn_global_load_async_to_lds_b128)
#define ASYNC_LDS 1
#endif
#endif

#if defined(__AMDGCN__) && defined(ASYNC_LDS)
#warning "gfx1250 device pass: async-to-LDS builtin ACTIVE"
#endif
#if defined(__AMDGCN__) && !defined(ASYNC_LDS)
#warning "gfx1250 device pass: async-to-LDS builtin MISSING - sync fallback"
#endif

static __device__ __forceinline__ void async_copy16(const _Float16* g, _Float16* l) {
#ifdef ASYNC_LDS
  __builtin_amdgcn_global_load_async_to_lds_b128(
      (__attribute__((address_space(1))) vi4*)(void*)g,
      (__attribute__((address_space(3))) vi4*)(void*)l, 0, 0);
#else
  *(h8*)l = *(const h8*)g;
#endif
}

static __device__ __forceinline__ void wait_async_all() {
#ifdef ASYNC_LDS
  asm volatile("s_wait_asynccnt 0" ::: "memory");
#endif
}

// ---------------- WMMA helpers (CDNA5 f16 16x16x32, f32 acc) ----------------

static __device__ __forceinline__ f8 wmma16(h16 a, h16 b, f8 c) {
  return __builtin_amdgcn_wmma_f32_16x16x32_f16(false, a, false, b, (short)0, c,
                                                false, false);
}

// A fragment: 16x32 f16, row-major source (row stride ld halves).
static __device__ __forceinline__ h16 load_a_frag(const _Float16* base, int ld,
                                                  int row, int lane) {
  const int grp = lane >> 4;
  const _Float16* p = base + row * ld + 8 * grp;
  h8 lo = *(const h8*)p;
  h8 hi = *(const h8*)(p + 16);
  h16 a;
#pragma unroll
  for (int i = 0; i < 8; ++i) { a[i] = lo[i]; a[8 + i] = hi[i]; }
  return a;
}

// B fragment: 32x16 f16, source stored K-contiguous per output column (Bt[n][k]).
static __device__ __forceinline__ h16 load_b_frag(const _Float16* base, int ld,
                                                  int col, int lane) {
  const int grp = lane >> 4;
  const _Float16* p = base + col * ld + 8 * grp;
  h8 lo = *(const h8*)p;
  h8 hi = *(const h8*)(p + 16);
  h16 b;
#pragma unroll
  for (int i = 0; i < 8; ++i) { b[i] = lo[i]; b[8 + i] = hi[i]; }
  return b;
}

// ---------------- elementwise f32 -> f16 (weights) ----------------

__global__ __launch_bounds__(256) void f32tof16_kernel(const float* __restrict__ s,
                                                       _Float16* __restrict__ d) {
  const size_t i = ((size_t)blockIdx.x * 256 + threadIdx.x) * 8;
#pragma unroll
  for (int j = 0; j < 8; ++j) d[i + j] = (_Float16)s[i + j];
}

// ---------------- LayerNorm (rows of 3072), fp32 in -> f16 out ----------------

__global__ __launch_bounds__(256) void ln3072_kernel(const float* __restrict__ src,
                                                     _Float16* __restrict__ dst) {
  const int row = blockIdx.x;
  const int tid = threadIdx.x;
  const float* x = src + (size_t)row * C_;
  float v[12];
  float s = 0.f;
#pragma unroll
  for (int i = 0; i < 12; ++i) { v[i] = x[tid + i * 256]; s += v[i]; }
  __shared__ float red[256];
  red[tid] = s;
  __syncthreads();
  for (int off = 128; off > 0; off >>= 1) {
    if (tid < off) red[tid] += red[tid + off];
    __syncthreads();
  }
  const float mu = red[0] * (1.f / C_);
  __syncthreads();
  float s2 = 0.f;
#pragma unroll
  for (int i = 0; i < 12; ++i) { float d = v[i] - mu; s2 += d * d; }
  red[tid] = s2;
  __syncthreads();
  for (int off = 128; off > 0; off >>= 1) {
    if (tid < off) red[tid] += red[tid + off];
    __syncthreads();
  }
  const float inv = rsqrtf(red[0] * (1.f / C_) + 1e-6f);
  _Float16* o = dst + (size_t)row * C_;
#pragma unroll
  for (int i = 0; i < 12; ++i) o[tid + i * 256] = (_Float16)((v[i] - mu) * inv);
}

// ---------------- Tiled WMMA GEMM:  out = A(f16) @ Wh(f16)^T + bias ----------------
// Block tile 128x256, K-tile 64, async double-buffered LDS.
// 8 waves in 2x4; each wave 64x64 via 4x4 WMMA accumulators.

template <bool OUT_HALF, bool USE_MASK>
__global__ __launch_bounds__(256) void gemm_wmma_kernel(
    const _Float16* __restrict__ A, const _Float16* __restrict__ Wh,
    const float* __restrict__ bias, float* __restrict__ outF,
    _Float16* __restrict__ outH, const float* __restrict__ mask, int M, int Nc,
    int Kc) {
  constexpr int BM = 128, BN = 256, BK = 64, LDT = 72;  // 144B rows: 16B aligned, odd*16
  __shared__ _Float16 As[2 * BM * LDT];
  __shared__ _Float16 Bs[2 * BN * LDT];
  const int tid = threadIdx.x;
  const int lane = tid & 31;
  const int wid = tid >> 5;
  const int wm = wid >> 2;  // 0..1
  const int wn = wid & 3;   // 0..3
  const int cl = lane & 15;
  const int grp = lane >> 4;
  const int row0 = blockIdx.y * BM;
  const int col0 = blockIdx.x * BN;

  f8 acc[4][4];
#pragma unroll
  for (int ni = 0; ni < 4; ++ni) {
    const float bv = bias[col0 + wn * 64 + ni * 16 + cl];
#pragma unroll
    for (int mi = 0; mi < 4; ++mi)
#pragma unroll
      for (int r = 0; r < 8; ++r) acc[mi][ni][r] = bv;
  }

  const int ar = tid >> 1;        // A row per thread (0..127)
  const int ak = (tid & 1) * 32;  // A k-chunk (0/32)

  auto stage = [&](int buf, int k0) {
    const _Float16* ga = A + (size_t)(row0 + ar) * Kc + k0 + ak;
    _Float16* la = As + buf * (BM * LDT) + ar * LDT + ak;
#pragma unroll
    for (int i = 0; i < 4; ++i) async_copy16(ga + i * 8, la + i * 8);
    const _Float16* gb = Wh + (size_t)(col0 + tid) * Kc + k0;
    _Float16* lb = Bs + buf * (BN * LDT) + tid * LDT;
#pragma unroll
    for (int i = 0; i < 8; ++i) async_copy16(gb + i * 8, lb + i * 8);
  };

  stage(0, 0);
  wait_async_all();
  __syncthreads();

  int buf = 0;
  for (int k0 = 0; k0 < Kc; k0 += BK) {
    if (k0 + BK < Kc) stage(buf ^ 1, k0 + BK);  // prefetch next stage (async)

    const _Float16* Ab = As + buf * (BM * LDT);
    const _Float16* Bb = Bs + buf * (BN * LDT);
#pragma unroll
    for (int kk = 0; kk < BK; kk += 32) {
      h16 af[4], bf[4];
#pragma unroll
      for (int mi = 0; mi < 4; ++mi)
        af[mi] = load_a_frag(Ab + kk, LDT, wm * 64 + mi * 16 + cl, lane);
#pragma unroll
      for (int ni = 0; ni < 4; ++ni)
        bf[ni] = load_b_frag(Bb + kk, LDT, wn * 64 + ni * 16 + cl, lane);
#pragma unroll
      for (int mi = 0; mi < 4; ++mi)
#pragma unroll
        for (int ni = 0; ni < 4; ++ni)
          acc[mi][ni] = wmma16(af[mi], bf[ni], acc[mi][ni]);
    }
    wait_async_all();
    __syncthreads();
    buf ^= 1;
  }

#pragma unroll
  for (int mi = 0; mi < 4; ++mi) {
#pragma unroll
    for (int r = 0; r < 8; ++r) {
      const int row = row0 + wm * 64 + mi * 16 + r + 8 * grp;
      float mval = 1.f;
      if constexpr (USE_MASK) mval = mask[row];
#pragma unroll
      for (int ni = 0; ni < 4; ++ni) {
        const int col = col0 + wn * 64 + ni * 16 + cl;
        const float v = acc[mi][ni][r] * mval;
        if constexpr (OUT_HALF)
          outH[(size_t)row * Nc + col] = (_Float16)v;
        else
          outF[(size_t)row * Nc + col] = v;
      }
    }
  }
}

// ---------------- RMSNorm (wave per group of 128) ----------------

__global__ __launch_bounds__(256) void rmsq_kernel(_Float16* __restrict__ q,
                                                   const float* __restrict__ w) {
  const int lane = threadIdx.x & 31;
  const int wid = threadIdx.x >> 5;
  const size_t g = (size_t)blockIdx.x * 8 + wid;
  _Float16* p = q + g * D_ + lane * 4;
  float v[4];
  float ss = 0.f;
#pragma unroll
  for (int i = 0; i < 4; ++i) { v[i] = (float)p[i]; ss += v[i] * v[i]; }
#pragma unroll
  for (int off = 16; off >= 1; off >>= 1) ss += __shfl_xor(ss, off, 32);
  const float inv = rsqrtf(ss * (1.f / D_) + 1e-6f);
#pragma unroll
  for (int i = 0; i < 4; ++i) p[i] = (_Float16)(v[i] * inv * w[lane * 4 + i]);
}

__global__ __launch_bounds__(256) void rmsk_kernel(const float* __restrict__ kv,
                                                   const float* __restrict__ w,
                                                   _Float16* __restrict__ kh) {
  const int lane = threadIdx.x & 31;
  const int wid = threadIdx.x >> 5;
  const int g = blockIdx.x * 8 + wid;  // row*24 + h
  const int row = g / H_;
  const int h = g % H_;
  const float* p = kv + (size_t)row * (2 * C_) + h * D_ + lane * 4;
  float v[4];
  float ss = 0.f;
#pragma unroll
  for (int i = 0; i < 4; ++i) { v[i] = p[i]; ss += v[i] * v[i]; }
#pragma unroll
  for (int off = 16; off >= 1; off >>= 1) ss += __shfl_xor(ss, off, 32);
  const float inv = rsqrtf(ss * (1.f / D_) + 1e-6f);
  _Float16* o = kh + (size_t)g * D_ + lane * 4;
#pragma unroll
  for (int i = 0; i < 4; ++i) o[i] = (_Float16)(v[i] * inv * w[lane * 4 + i]);
}

__global__ __launch_bounds__(256) void vconv_kernel(const float* __restrict__ kv,
                                                    _Float16* __restrict__ vh) {
  const size_t i = (size_t)blockIdx.x * 256 + threadIdx.x;
  const size_t row = i / C_;
  const size_t col = i - row * C_;
  vh[i] = (_Float16)kv[row * (2 * C_) + C_ + col];
}

// ---------------- Attention: one block per (b,t,h) ----------------

__global__ __launch_bounds__(256) void attn_kernel(const _Float16* __restrict__ qh,
                                                   const _Float16* __restrict__ kh,
                                                   const _Float16* __restrict__ vh,
                                                   _Float16* __restrict__ attn) {
  const int bi = blockIdx.x;
  const int h = bi % H_;
  const int bt = bi / H_;
  const int tid = threadIdx.x;
  const int lane = tid & 31;
  const int wid = tid >> 5;
  const int cl = lane & 15;
  const int grp = lane >> 4;

  __shared__ _Float16 klds[NM_ * D_];  // [tok][d]
  __shared__ _Float16 vt[D_ * NM_];    // [d][tok]
  __shared__ float sco[S_ * NM_];
  __shared__ _Float16 prb[S_ * NM_];

  {  // stage k
    const int tok = tid >> 3;
    const int d0 = (tid & 7) * 16;
    const _Float16* g = kh + (size_t)(bt * NM_ + tok) * C_ + h * D_ + d0;
    _Float16* l = klds + tok * D_ + d0;
    *(h8*)l = *(const h8*)g;
    *(h8*)(l + 8) = *(const h8*)(g + 8);
  }
#pragma unroll
  for (int j = 0; j < 16; ++j) {  // stage v transposed
    const int li = tid * 16 + j;
    const int d = li >> 5;
    const int tok = li & 31;
    vt[d * NM_ + tok] = vh[(size_t)(bt * NM_ + tok) * C_ + h * D_ + d];
  }
  __syncthreads();

  f8 acc[2][2];
#pragma unroll
  for (int mi = 0; mi < 2; ++mi)
#pragma unroll
    for (int ni = 0; ni < 2; ++ni)
#pragma unroll
      for (int r = 0; r < 8; ++r) acc[mi][ni][r] = 0.f;

  const _Float16* qbase = qh + (size_t)bt * S_ * C_ + h * D_;
#pragma unroll
  for (int ks = 0; ks < D_; ks += 32) {
    h16 aq[2], bk[2];
#pragma unroll
    for (int mi = 0; mi < 2; ++mi)
      aq[mi] = load_a_frag(qbase + ks, C_, wid * 32 + mi * 16 + cl, lane);
#pragma unroll
    for (int ni = 0; ni < 2; ++ni)
      bk[ni] = load_b_frag(klds + ks, D_, ni * 16 + cl, lane);
#pragma unroll
    for (int mi = 0; mi < 2; ++mi)
#pragma unroll
      for (int ni = 0; ni < 2; ++ni)
        acc[mi][ni] = wmma16(aq[mi], bk[ni], acc[mi][ni]);
  }
  const float scale = 0.08838834764831845f;  // 1/sqrt(128)
#pragma unroll
  for (int mi = 0; mi < 2; ++mi)
#pragma unroll
    for (int ni = 0; ni < 2; ++ni)
#pragma unroll
      for (int r = 0; r < 8; ++r) {
        const int row = wid * 32 + mi * 16 + r + 8 * grp;
        sco[row * NM_ + ni * 16 + cl] = acc[mi][ni][r] * scale;
      }
  __syncthreads();

  {  // softmax, one row per thread
    float e[NM_];
    float mx = -1e30f;
    const float* sr = sco + tid * NM_;
#pragma unroll
    for (int i = 0; i < NM_; ++i) { e[i] = sr[i]; mx = fmaxf(mx, e[i]); }
    float sum = 0.f;
#pragma unroll
    for (int i = 0; i < NM_; ++i) { e[i] = __expf(e[i] - mx); sum += e[i]; }
    const float inv = 1.f / sum;
    _Float16* pr = prb + tid * NM_;
#pragma unroll
    for (int i = 0; i < NM_; ++i) pr[i] = (_Float16)(e[i] * inv);
  }
  __syncthreads();

  h16 ap[2];
#pragma unroll
  for (int mi = 0; mi < 2; ++mi)
    ap[mi] = load_a_frag(prb, NM_, wid * 32 + mi * 16 + cl, lane);
  _Float16* obase = attn + (size_t)bt * S_ * C_ + h * D_;
#pragma unroll
  for (int ni = 0; ni < 8; ++ni) {
    h16 bv = load_b_frag(vt, NM_, ni * 16 + cl, lane);
#pragma unroll
    for (int mi = 0; mi < 2; ++mi) {
      f8 a = {0.f, 0.f, 0.f, 0.f, 0.f, 0.f, 0.f, 0.f};
      a = wmma16(ap[mi], bv, a);
#pragma unroll
      for (int r = 0; r < 8; ++r) {
        const int row = wid * 32 + mi * 16 + r + 8 * grp;
        obase[(size_t)row * C_ + ni * 16 + cl] = (_Float16)a[r];
      }
    }
  }
}

// ---------------- Launcher ----------------

extern "C" void kernel_launch(void* const* d_in, const int* in_sizes, int n_in,
                              void* d_out, int out_size, void* d_ws, size_t ws_size,
                              hipStream_t stream) {
  (void)in_sizes; (void)n_in; (void)out_size; (void)ws_size;
  const float* x    = (const float*)d_in[0];
  const float* mv   = (const float*)d_in[1];
  const float* mask = (const float*)d_in[2];
  const float* w_kv = (const float*)d_in[3];
  const float* b_kv = (const float*)d_in[4];
  const float* w_q  = (const float*)d_in[5];
  const float* b_q  = (const float*)d_in[6];
  const float* w_o  = (const float*)d_in[7];
  const float* b_o  = (const float*)d_in[8];
  const float* qw   = (const float*)d_in[9];
  const float* kw   = (const float*)d_in[10];
  float* out = (float*)d_out;

  const size_t ROWS_X = (size_t)B_ * T_ * S_;   // 8192
  const size_t ROWS_M = (size_t)B_ * T_ * NM_;  // 1024

  char* ws = (char*)d_ws;
  _Float16* xf   = (_Float16*)ws; ws += ROWS_X * C_ * 2;      // ln(x); reused as attn out
  _Float16* xm   = (_Float16*)ws; ws += ROWS_M * C_ * 2;      // ln(motion)
  float*    kv   = (float*)ws;    ws += ROWS_M * 2 * C_ * 4;  // kv GEMM out (f32)
  _Float16* qh   = (_Float16*)ws; ws += ROWS_X * C_ * 2;      // q f16 (rmsnorm in place)
  _Float16* kh   = (_Float16*)ws; ws += ROWS_M * C_ * 2;      // k normalized f16
  _Float16* vh   = (_Float16*)ws; ws += ROWS_M * C_ * 2;      // v f16
  _Float16* wkvh = (_Float16*)ws; ws += (size_t)2 * C_ * C_ * 2;
  _Float16* wqh  = (_Float16*)ws; ws += (size_t)C_ * C_ * 2;
  _Float16* woh  = (_Float16*)ws; ws += (size_t)C_ * C_ * 2;
  _Float16* at   = xf;  // alias: xf dead after q GEMM

  // weight conversion f32 -> f16 (one-shot, memory bound)
  f32tof16_kernel<<<(unsigned)((size_t)2 * C_ * C_ / 2048), 256, 0, stream>>>(w_kv, wkvh);
  f32tof16_kernel<<<(unsigned)((size_t)C_ * C_ / 2048), 256, 0, stream>>>(w_q, wqh);
  f32tof16_kernel<<<(unsigned)((size_t)C_ * C_ / 2048), 256, 0, stream>>>(w_o, woh);

  ln3072_kernel<<<(unsigned)ROWS_X, 256, 0, stream>>>(x, xf);
  ln3072_kernel<<<(unsigned)ROWS_M, 256, 0, stream>>>(mv, xm);

  gemm_wmma_kernel<false, false><<<dim3(2 * C_ / 256, ROWS_M / 128), 256, 0, stream>>>(
      xm, wkvh, b_kv, kv, nullptr, nullptr, (int)ROWS_M, 2 * C_, C_);
  gemm_wmma_kernel<true, false><<<dim3(C_ / 256, ROWS_X / 128), 256, 0, stream>>>(
      xf, wqh, b_q, nullptr, qh, nullptr, (int)ROWS_X, C_, C_);

  rmsq_kernel<<<(unsigned)(ROWS_X * H_ / 8), 256, 0, stream>>>(qh, qw);
  rmsk_kernel<<<(unsigned)(ROWS_M * H_ / 8), 256, 0, stream>>>(kv, kw, kh);
  vconv_kernel<<<(unsigned)(ROWS_M * C_ / 256), 256, 0, stream>>>(kv, vh);

  attn_kernel<<<B_ * T_ * H_, 256, 0, stream>>>(qh, kh, vh, at);

  gemm_wmma_kernel<false, true><<<dim3(C_ / 256, ROWS_X / 128), 256, 0, stream>>>(
      at, woh, b_o, out, nullptr, mask, (int)ROWS_X, C_, C_);
}
